// TypedCrossInteraction_82214263980113
// MI455X (gfx1250) — compile-verified, matrix operand order checked by reading, not run on hardware
//
#include <hip/hip_runtime.h>
#include <math.h>

typedef _Float16 h16;
typedef __attribute__((ext_vector_type(16))) _Float16 v16h;
typedef __attribute__((ext_vector_type(8)))  _Float16 v8h;
typedef __attribute__((ext_vector_type(8)))  float    v8f;

#define H        128
#define NRAD     24
#define KEXP     4
#define EDGE_DIM 353
#define KPAD     384
#define KC1      12      /* 384/32 K-chunks for layer-1 */
#define KC2      4       /* 128/32 K-chunks */
#define EPB      32      /* edges per block (2 WMMA tiles) */
#define PI_F     3.14159265358979f
#define CUTOFF_F 4.5f

/* ---- packed f16 weight layout (offsets in halves) ---- */
#define SZ_W1   (8*12*32*16)   /* [353->128] : 8 ntiles * 12 kc   */
#define SZ_W2   (8*4*32*16)    /* [128->128] : 8 ntiles * 4 kc    */
#define SZ_GW2  (1*4*32*16)    /* [128->4pad16]                   */
#define SZ_CW1  (4*4*32*16)    /* [128->64]                       */
#define PK_GW1   0
#define PK_EW1   (PK_GW1 + SZ_W1)
#define PK_EW2   (PK_EW1 + 4*SZ_W1)
#define PK_GW2   (PK_EW2 + 4*SZ_W2)
#define PK_CSW1  (PK_GW2 + SZ_GW2)
#define PK_CVW1  (PK_CSW1 + SZ_CW1)
#define PK_TOTAL (PK_CVW1 + SZ_CW1)

__device__ __forceinline__ float siluf(float x) { return x / (1.0f + __expf(-x)); }
__device__ __forceinline__ float softplusf(float x) {
  return fmaxf(x, 0.0f) + log1pf(__expf(-fabsf(x)));
}
__device__ __forceinline__ v8f zero8() {
  v8f z;
#pragma unroll
  for (int i = 0; i < 8; ++i) z[i] = 0.0f;
  return z;
}
/* A-matrix f16 lane layout for V_WMMA_F32_16X16X32_F16:
   lane L: row = L&15, half h = L>>4; element i holds K = kc*32 + h*8 + (i<8 ? i : i+8).
   -> two contiguous 8-half (16B) runs per chunk. */
__device__ __forceinline__ v16h load_a16(const h16* row, int kc, int hf) {
  const v8h* p0 = (const v8h*)(row + kc * 32 + hf * 8);
  const v8h* p1 = (const v8h*)(row + kc * 32 + 16 + hf * 8);
  v8h lo = *p0, hi = *p1;
  v16h a;
#pragma unroll
  for (int i = 0; i < 8; ++i) { a[i] = lo[i]; a[i + 8] = hi[i]; }
  return a;
}
__device__ __forceinline__ v16h load_a32(const float* row, int kc, int hf) {
  v16h a;
#pragma unroll
  for (int i = 0; i < 8; ++i) {
    a[i]     = (h16)row[kc * 32 + hf * 8 + i];
    a[i + 8] = (h16)row[kc * 32 + 16 + hf * 8 + i];
  }
  return a;
}
#define WMMA(acc, a, b) \
  acc = __builtin_amdgcn_wmma_f32_16x16x32_f16(false, a, false, b, (short)0, acc, false, false)

/* ------------------------------------------------------------------ */
__global__ void zero_f32(float* p, int n) {
  for (int i = blockIdx.x * blockDim.x + threadIdx.x; i < n; i += gridDim.x * blockDim.x)
    p[i] = 0.0f;
}

/* pack row-major W[Kdim][Ndim] (fp32) into WMMA B lane layout (f16) */
__global__ void pack_w(const float* w, h16* dst, int Kdim, int Ndim, int Ntiles, int KC) {
  int idx = blockIdx.x * blockDim.x + threadIdx.x;
  int total = Ntiles * KC * 32;
  if (idx >= total) return;
  int lane = idx & 31;
  int kc   = (idx >> 5) % KC;
  int nt   = idx / (32 * KC);
  int n    = nt * 16 + (lane & 15);
  int hf   = lane >> 4;
#pragma unroll
  for (int i = 0; i < 16; ++i) {
    int k = kc * 32 + hf * 8 + (i < 8 ? i : i + 8);
    float v = (k < Kdim && n < Ndim) ? w[k * Ndim + n] : 0.0f;
    dst[idx * 16 + i] = (h16)v;
  }
}

/* per-node: lin = h@msg_w+b ; lj = softplus(mlp2(h, phys)) + 1e-3 */
__global__ __launch_bounds__(128) void node_pre(
    const float* h, const float* pw1, const float* pb1, const float* pw2, const float* pb2,
    const float* mw, const float* mb, float* lin, float* lj) {
  int n = blockIdx.x, tid = threadIdx.x;
  __shared__ float hh[H];
  __shared__ float hd[H];
  hh[tid] = h[n * H + tid];
  __syncthreads();
  float a1 = pb1[tid], al = mb[tid];
  for (int r = 0; r < H; ++r) { float x = hh[r]; a1 += x * pw1[r * H + tid]; al += x * mw[r * H + tid]; }
  hd[tid] = siluf(a1);
  lin[n * H + tid] = al;
  __syncthreads();
  if (tid < 2) {
    float a2 = pb2[tid];
    for (int r = 0; r < H; ++r) a2 += hd[r] * pw2[r * 2 + tid];
    lj[n * 2 + tid] = softplusf(a2) + 1e-3f;
  }
}

/* per-node finalize: h_out = h + LN(silu([h,agg/c]@W+b)); p_out = p + 0.08*d/c */
__global__ __launch_bounds__(128) void node_finalize(
    const float* h, const float* pin, const float* agg, const float* cnt, const float* dsum,
    const float* updw, const float* updb, const float* lng, const float* lnb,
    float* hout, float* pout) {
  int n = blockIdx.x, tid = threadIdx.x;
  __shared__ float cat[2 * H];
  __shared__ float red[H];
  float c = fmaxf(cnt[n], 1.0f);
  cat[tid]     = h[n * H + tid];
  cat[H + tid] = agg[n * H + tid] / c;
  __syncthreads();
  float u = updb[tid];
  for (int r = 0; r < 2 * H; ++r) u += cat[r] * updw[r * H + tid];
  u = siluf(u);
  red[tid] = u;
  __syncthreads();
  for (int s = 64; s > 0; s >>= 1) { if (tid < s) red[tid] += red[tid + s]; __syncthreads(); }
  float mean = red[0] / (float)H;
  __syncthreads();
  red[tid] = (u - mean) * (u - mean);
  __syncthreads();
  for (int s = 64; s > 0; s >>= 1) { if (tid < s) red[tid] += red[tid + s]; __syncthreads(); }
  float var = red[0] / (float)H;
  hout[n * H + tid] = h[n * H + tid] + (u - mean) * rsqrtf(var + 1e-5f) * lng[tid] + lnb[tid];
  if (tid < 3) pout[n * 3 + tid] = pin[n * 3 + tid] + 0.08f * dsum[n * 3 + tid] / c;
}

/* ------------------------------------------------------------------ */
struct EdgeArgs {
  const int *idxS, *idxV;
  const float *h_s, *h_v, *p_s, *p_v, *q_s, *q_v;
  const float *role_s, *role_v, *normal_s, *normal_v, *dipole_s, *dipole_v;
  const float *lj_s, *lj_v;
  const float *type_w1, *type_b1, *type_w2, *type_b2;
  const float *norm_g, *norm_b;
  const h16* pk;
  const float *gate_b1, *gate_b2, *exp_b1, *exp_b2;
  const float *cs_b1, *cs_w2, *cs_b2, *cv_b1, *cv_w2, *cv_b2;
  const float *lin_s, *lin_v;
  float *aggS, *aggV, *cntS, *cntV, *dS, *dV;
  int E;
};

__global__ __launch_bounds__(256) void edge_kernel(EdgeArgs args) {
  __shared__ __align__(16) h16 es[EPB][KPAD];     /* 24 KB : layernormed features (f16) */
  __shared__ __align__(16) h16 sh_hid[EPB][H];    /*  8 KB : gate/expert hidden         */
  __shared__ float sh_eh[EPB][H];                 /* 16 KB : edge_hidden (f32)          */
  __shared__ __align__(16) h16 sh_ch[2][EPB][64]; /*  8 KB : coord hidden               */
  __shared__ float gatev[EPB][4];
  __shared__ float mean_s[EPB], istd_s[EPB];
  __shared__ float th1[8][64];
  __shared__ float unitL[EPB][3];
  __shared__ float coefL[2][EPB];
  __shared__ int   isL[EPB], ivL[EPB];

  const int tid  = threadIdx.x;
  const int lane = tid & 31;
  const int w    = tid >> 5;      /* wave id 0..7 */
  const int col  = lane & 15;
  const int hf   = lane >> 4;
  const int e0   = blockIdx.x * EPB;

  /* ---------------- phase 1: build raw edge features (f16) ---------------- */
  for (int ee = w; ee < EPB; ee += 8) {
    int eg = e0 + ee;
    if (eg >= args.E) eg = args.E - 1;
    int is = args.idxS[eg], iv = args.idxV[eg];
    if (lane == 0) { isL[ee] = is; ivL[ee] = iv; }
    for (int c = lane; c < H; c += 32) {
      es[ee][c]     = (h16)args.h_s[is * H + c];
      es[ee][H + c] = (h16)args.h_v[iv * H + c];
    }
    float rx = args.p_s[is * 3 + 0] - args.p_v[iv * 3 + 0];
    float ry = args.p_s[is * 3 + 1] - args.p_v[iv * 3 + 1];
    float rz = args.p_s[is * 3 + 2] - args.p_v[iv * 3 + 2];
    float dist = sqrtf(rx * rx + ry * ry + rz * rz + 1e-8f);
    float safe = fmaxf(dist, 0.5f);
    float clip = fmaxf(dist, 1e-6f);
    if (lane < NRAD) {
      float f = (float)(lane + 1) * (PI_F / CUTOFF_F);
      es[ee][256 + lane] = (h16)(__sinf(f * clip) / clip);
    }
    float qs = args.q_s[is], qv = args.q_v[iv];
    float qe = qs * qv, cdiff = fabsf(qs - qv), e_coul = qe / safe;
    float epsv  = sqrtf(args.lj_s[is * 2] * args.lj_v[iv * 2] + 1e-8f);
    float sig   = 0.5f * (args.lj_s[is * 2 + 1] + args.lj_v[iv * 2 + 1]);
    float ratio = fminf(sig / safe, 1.5f);
    float r2 = ratio * ratio, r6 = r2 * r2 * r2;
    float e_lj = 4.0f * epsv * (r6 * r6 - r6);
    float rs0 = args.role_s[is * 5 + 0], rs1 = args.role_s[is * 5 + 1], rs2 = args.role_s[is * 5 + 2];
    float rs3 = args.role_s[is * 5 + 3], rs4 = args.role_s[is * 5 + 4];
    float rv0 = args.role_v[iv * 5 + 0], rv1 = args.role_v[iv * 5 + 1], rv2 = args.role_v[iv * 5 + 2];
    float rv3 = args.role_v[iv * 5 + 3], rv4 = args.role_v[iv * 5 + 4];
    float da = rs0 * rv1 + rs1 * rv0;
    float ap = rs2 * rv2;
    float nd = args.normal_s[is * 3] * args.normal_v[iv * 3]
             + args.normal_s[is * 3 + 1] * args.normal_v[iv * 3 + 1]
             + args.normal_s[is * 3 + 2] * args.normal_v[iv * 3 + 2];
    float na  = ap * fabsf(nd);
    float dal = args.dipole_s[is * 3] * args.dipole_v[iv * 3]
              + args.dipole_s[is * 3 + 1] * args.dipole_v[iv * 3 + 1]
              + args.dipole_s[is * 3 + 2] * args.dipole_v[iv * 3 + 2];
    float dop = 1.0f - fabsf(dal);
    float hp = rs3 * rv3, pp = rs4 * rv4, hpp = rs3 * rv4 + rs4 * rv3;
    float it[8] = {da, ap, na, dal, dop, hp, pp, hpp};
    /* tiny type MLP: layer1 -> per-wave LDS, intra-wave wait, layer2 */
#pragma unroll
    for (int j = 0; j < 2; ++j) {
      int c = 2 * lane + j;
      float t = args.type_b1[c];
#pragma unroll
      for (int i = 0; i < 8; ++i) t += it[i] * args.type_w1[i * 64 + c];
      th1[w][c] = siluf(t);
    }
    asm volatile("s_wait_dscnt 0" ::: "memory");   /* intra-wave LDS RAW */
#pragma unroll
    for (int j = 0; j < 2; ++j) {
      int c = 2 * lane + j;
      float t = args.type_b2[c];
      for (int r = 0; r < 64; ++r) t += th1[w][r] * args.type_w2[r * 64 + c];
      es[ee][289 + c] = (h16)t;
    }
    if (lane == 0) {
      es[ee][280] = (h16)(dist / CUTOFF_F);
      es[ee][281] = (h16)(1.0f / safe);
      es[ee][282] = (h16)qe;
      es[ee][283] = (h16)cdiff;
      es[ee][284] = (h16)e_lj;
      es[ee][285] = (h16)e_coul;
      es[ee][286] = (h16)da;
      es[ee][287] = (h16)na;
      es[ee][288] = (h16)hp;
      unitL[ee][0] = rx / safe; unitL[ee][1] = ry / safe; unitL[ee][2] = rz / safe;
    }
  }
  /* zero K padding 353..383 */
  for (int idx = tid; idx < EPB * (KPAD - EDGE_DIM); idx += 256) {
    es[idx / (KPAD - EDGE_DIM)][EDGE_DIM + idx % (KPAD - EDGE_DIM)] = (h16)0.0f;
  }
  __syncthreads();

  /* ---------------- phase 2: layernorm (wave reductions) ---------------- */
  for (int ee = w; ee < EPB; ee += 8) {
    float s = 0.0f, s2 = 0.0f;
    for (int d = lane; d < EDGE_DIM; d += 32) { float x = (float)es[ee][d]; s += x; s2 += x * x; }
#pragma unroll
    for (int off = 16; off > 0; off >>= 1) { s += __shfl_xor(s, off); s2 += __shfl_xor(s2, off); }
    if (lane == 0) {
      float m = s / (float)EDGE_DIM;
      float v = s2 / (float)EDGE_DIM - m * m;
      mean_s[ee] = m;
      istd_s[ee] = rsqrtf(fmaxf(v, 0.0f) + 1e-5f);
    }
  }
  __syncthreads();
  for (int idx = tid; idx < EPB * EDGE_DIM; idx += 256) {
    int e = idx / EDGE_DIM, d = idx % EDGE_DIM;
    float x = (float)es[e][d];
    es[e][d] = (h16)((x - mean_s[e]) * istd_s[e] * args.norm_g[d] + args.norm_b[d]);
  }
  __syncthreads();

  /* ---------------- phase 3: gate hidden (WMMA, ntile = wave) ---------------- */
  {
    v8f acc0 = zero8(), acc1 = zero8();
    for (int kc = 0; kc < KC1; ++kc) {
      v16h b  = *(const v16h*)(args.pk + PK_GW1 + (size_t)((w * KC1 + kc) * 32 + lane) * 16);
      v16h a0 = load_a16(&es[col][0], kc, hf);
      v16h a1 = load_a16(&es[16 + col][0], kc, hf);
      WMMA(acc0, a0, b);
      WMMA(acc1, a1, b);
    }
    float bias = args.gate_b1[w * 16 + col];
#pragma unroll
    for (int r = 0; r < 8; ++r) {
      sh_hid[r + hf * 8][w * 16 + col]      = (h16)siluf(acc0[r] + bias);
      sh_hid[16 + r + hf * 8][w * 16 + col] = (h16)siluf(acc1[r] + bias);
    }
  }
  __syncthreads();

  /* ---------------- phase 4: gate logits + softmax ---------------- */
  if (w == 0) {
    v8f acc0 = zero8(), acc1 = zero8();
    for (int kc = 0; kc < KC2; ++kc) {
      v16h b  = *(const v16h*)(args.pk + PK_GW2 + (size_t)(kc * 32 + lane) * 16);
      v16h a0 = load_a16(&sh_hid[col][0], kc, hf);
      v16h a1 = load_a16(&sh_hid[16 + col][0], kc, hf);
      WMMA(acc0, a0, b);
      WMMA(acc1, a1, b);
    }
    if (col < 4) {
      float gb = args.gate_b2[col];
#pragma unroll
      for (int r = 0; r < 8; ++r) {
        gatev[r + hf * 8][col]      = acc0[r] + gb;
        gatev[16 + r + hf * 8][col] = acc1[r] + gb;
      }
    }
  }
  __syncthreads();
  if (tid < EPB) {
    float x0 = gatev[tid][0], x1 = gatev[tid][1], x2 = gatev[tid][2], x3 = gatev[tid][3];
    float m  = fmaxf(fmaxf(x0, x1), fmaxf(x2, x3));
    float g0 = __expf(x0 - m), g1 = __expf(x1 - m), g2 = __expf(x2 - m), g3 = __expf(x3 - m);
    float s  = g0 + g1 + g2 + g3;
    gatev[tid][0] = g0 / s; gatev[tid][1] = g1 / s; gatev[tid][2] = g2 / s; gatev[tid][3] = g3 / s;
  }
  __syncthreads();

  /* ---------------- phase 5: mixture of experts (WMMA) ---------------- */
  float eha0[8], eha1[8];
#pragma unroll
  for (int r = 0; r < 8; ++r) { eha0[r] = 0.0f; eha1[r] = 0.0f; }
  for (int k = 0; k < KEXP; ++k) {
    {
      const h16* W1 = args.pk + PK_EW1 + (size_t)k * SZ_W1;
      v8f acc0 = zero8(), acc1 = zero8();
      for (int kc = 0; kc < KC1; ++kc) {
        v16h b  = *(const v16h*)(W1 + (size_t)((w * KC1 + kc) * 32 + lane) * 16);
        v16h a0 = load_a16(&es[col][0], kc, hf);
        v16h a1 = load_a16(&es[16 + col][0], kc, hf);
        WMMA(acc0, a0, b);
        WMMA(acc1, a1, b);
      }
      float b1v = args.exp_b1[k * H + w * 16 + col];
#pragma unroll
      for (int r = 0; r < 8; ++r) {
        sh_hid[r + hf * 8][w * 16 + col]      = (h16)siluf(acc0[r] + b1v);
        sh_hid[16 + r + hf * 8][w * 16 + col] = (h16)siluf(acc1[r] + b1v);
      }
    }
    __syncthreads();
    {
      const h16* W2 = args.pk + PK_EW2 + (size_t)k * SZ_W2;
      v8f acc0 = zero8(), acc1 = zero8();
      for (int kc = 0; kc < KC2; ++kc) {
        v16h b  = *(const v16h*)(W2 + (size_t)((w * KC2 + kc) * 32 + lane) * 16);
        v16h a0 = load_a16(&sh_hid[col][0], kc, hf);
        v16h a1 = load_a16(&sh_hid[16 + col][0], kc, hf);
        WMMA(acc0, a0, b);
        WMMA(acc1, a1, b);
      }
      float b2v = args.exp_b2[k * H + w * 16 + col];
#pragma unroll
      for (int r = 0; r < 8; ++r) {
        eha0[r] += gatev[r + hf * 8][k]      * siluf(acc0[r] + b2v);
        eha1[r] += gatev[16 + r + hf * 8][k] * siluf(acc1[r] + b2v);
      }
    }
    __syncthreads();
  }
#pragma unroll
  for (int r = 0; r < 8; ++r) {
    sh_eh[r + hf * 8][w * 16 + col]      = eha0[r];
    sh_eh[16 + r + hf * 8][w * 16 + col] = eha1[r];
  }
  __syncthreads();

  /* ---------------- phase 6: coord MLPs ---------------- */
  {
    int side = w >> 2, nt = w & 3;
    const h16*   Wc  = args.pk + (side ? PK_CVW1 : PK_CSW1);
    const float* bcp = side ? args.cv_b1 : args.cs_b1;
    v8f acc0 = zero8(), acc1 = zero8();
    for (int kc = 0; kc < KC2; ++kc) {
      v16h b  = *(const v16h*)(Wc + (size_t)((nt * KC2 + kc) * 32 + lane) * 16);
      v16h a0 = load_a32(&sh_eh[col][0], kc, hf);
      v16h a1 = load_a32(&sh_eh[16 + col][0], kc, hf);
      WMMA(acc0, a0, b);
      WMMA(acc1, a1, b);
    }
    float bb = bcp[nt * 16 + col];
#pragma unroll
    for (int r = 0; r < 8; ++r) {
      sh_ch[side][r + hf * 8][nt * 16 + col]      = (h16)siluf(acc0[r] + bb);
      sh_ch[side][16 + r + hf * 8][nt * 16 + col] = (h16)siluf(acc1[r] + bb);
    }
  }
  __syncthreads();
  if (tid < 2 * EPB) {
    int side = tid >> 5, e = tid & 31;
    const float* w2 = side ? args.cv_w2 : args.cs_w2;
    float sc = side ? args.cv_b2[0] : args.cs_b2[0];
    for (int j = 0; j < 64; ++j) sc += (float)sh_ch[side][e][j] * w2[j];
    coefL[side][e] = sc;
  }
  __syncthreads();

  /* ---------------- phase 7: scatter (atomics) ---------------- */
  for (int ee = w * 4; ee < w * 4 + 4; ++ee) {
    if (e0 + ee >= args.E) break;
    int is = isL[ee], iv = ivL[ee];
    for (int c = lane; c < H; c += 32) {
      float ev = sh_eh[ee][c];
      atomicAdd(&args.aggS[is * H + c], args.lin_v[iv * H + c] * ev);
      atomicAdd(&args.aggV[iv * H + c], args.lin_s[is * H + c] * ev);
    }
    if (lane == 0) { atomicAdd(&args.cntS[is], 1.0f); atomicAdd(&args.cntV[iv], 1.0f); }
    if (lane < 3) {
      atomicAdd(&args.dS[is * 3 + lane],  unitL[ee][lane] * coefL[0][ee]);
      atomicAdd(&args.dV[iv * 3 + lane], -unitL[ee][lane] * coefL[1][ee]);
    }
  }
}

/* ------------------------------------------------------------------ */
extern "C" void kernel_launch(void* const* d_in, const int* in_sizes, int n_in,
                              void* d_out, int out_size, void* d_ws, size_t ws_size,
                              hipStream_t stream) {
  const float* h_s = (const float*)d_in[0];
  const float* p_s = (const float*)d_in[1];
  const float* h_v = (const float*)d_in[3];
  const float* p_v = (const float*)d_in[4];
  const float* q_s = (const float*)d_in[6];
  const float* q_v = (const float*)d_in[7];
  const float* role_s   = (const float*)d_in[8];
  const float* role_v   = (const float*)d_in[9];
  const float* normal_s = (const float*)d_in[10];
  const float* normal_v = (const float*)d_in[11];
  const float* dipole_s = (const float*)d_in[12];
  const float* dipole_v = (const float*)d_in[13];
  const int*   cei      = (const int*)d_in[14];
  auto W = [&](int i) { return (const float*)d_in[i]; };

  const int NSn = in_sizes[0] / H;
  const int NVn = in_sizes[3] / H;
  const int E   = in_sizes[14] / 2;
  const int* idxS = cei;
  const int* idxV = cei + E;

  /* workspace layout */
  h16*   pk  = (h16*)d_ws;
  float* fws = (float*)((char*)d_ws + (size_t)((PK_TOTAL * 2 + 255) / 256) * 256);
  float* lj_s = fws;               float* lj_v = lj_s + (size_t)NSn * 2;
  float* lin_s = lj_v + (size_t)NVn * 2;  float* lin_v = lin_s + (size_t)NSn * H;
  float* aggS = lin_v + (size_t)NVn * H;  float* aggV = aggS + (size_t)NSn * H;
  float* cntS = aggV + (size_t)NVn * H;   float* cntV = cntS + NSn;
  float* dS   = cntV + NVn;               float* dV   = dS + (size_t)NSn * 3;

  /* zero atomic accumulators (agg, cnt, delta) — contiguous from aggS */
  int zeroN = NSn * H + NVn * H + NSn + NVn + NSn * 3 + NVn * 3;
  hipLaunchKernelGGL(zero_f32, dim3(1024), dim3(256), 0, stream, aggS, zeroN);

  /* pack shared weights to f16 B-layout */
  auto packLaunch = [&](const float* src, h16* dst, int Kdim, int Ndim, int Nt, int KC) {
    int total = Nt * KC * 32;
    hipLaunchKernelGGL(pack_w, dim3((total + 255) / 256), dim3(256), 0, stream,
                       src, dst, Kdim, Ndim, Nt, KC);
  };
  packLaunch(W(29), pk + PK_GW1, EDGE_DIM, H, 8, KC1);                       /* gate_w1 */
  for (int k = 0; k < KEXP; ++k)
    packLaunch(W(25) + (size_t)k * EDGE_DIM * H, pk + PK_EW1 + (size_t)k * SZ_W1,
               EDGE_DIM, H, 8, KC1);                                         /* exp_w1 */
  for (int k = 0; k < KEXP; ++k)
    packLaunch(W(27) + (size_t)k * H * H, pk + PK_EW2 + (size_t)k * SZ_W2,
               H, H, 8, KC2);                                                /* exp_w2 */
  packLaunch(W(31), pk + PK_GW2, H, 4, 1, KC2);                              /* gate_w2 */
  packLaunch(W(45), pk + PK_CSW1, H, 64, 4, KC2);                            /* cs_w1  */
  packLaunch(W(49), pk + PK_CVW1, H, 64, 4, KC2);                            /* cv_w1  */

  /* per-node pre-linears */
  hipLaunchKernelGGL(node_pre, dim3(NSn), dim3(128), 0, stream,
                     h_s, W(15), W(16), W(17), W(18), W(33), W(34), lin_s, lj_s);
  hipLaunchKernelGGL(node_pre, dim3(NVn), dim3(128), 0, stream,
                     h_v, W(15), W(16), W(17), W(18), W(35), W(36), lin_v, lj_v);

  /* edge kernel */
  EdgeArgs ea;
  ea.idxS = idxS; ea.idxV = idxV;
  ea.h_s = h_s; ea.h_v = h_v; ea.p_s = p_s; ea.p_v = p_v; ea.q_s = q_s; ea.q_v = q_v;
  ea.role_s = role_s; ea.role_v = role_v; ea.normal_s = normal_s; ea.normal_v = normal_v;
  ea.dipole_s = dipole_s; ea.dipole_v = dipole_v;
  ea.lj_s = lj_s; ea.lj_v = lj_v;
  ea.type_w1 = W(19); ea.type_b1 = W(20); ea.type_w2 = W(21); ea.type_b2 = W(22);
  ea.norm_g = W(23); ea.norm_b = W(24);
  ea.pk = pk;
  ea.gate_b1 = W(30); ea.gate_b2 = W(32); ea.exp_b1 = W(26); ea.exp_b2 = W(28);
  ea.cs_b1 = W(46); ea.cs_w2 = W(47); ea.cs_b2 = W(48);
  ea.cv_b1 = W(50); ea.cv_w2 = W(51); ea.cv_b2 = W(52);
  ea.lin_s = lin_s; ea.lin_v = lin_v;
  ea.aggS = aggS; ea.aggV = aggV; ea.cntS = cntS; ea.cntV = cntV; ea.dS = dS; ea.dV = dV;
  ea.E = E;
  hipLaunchKernelGGL(edge_kernel, dim3((E + EPB - 1) / EPB), dim3(256), 0, stream, ea);

  /* finalize: out = [h_s_out | p_s_out | h_v_out | p_v_out] */
  float* out = (float*)d_out;
  float* hs_out = out;
  float* ps_out = hs_out + (size_t)NSn * H;
  float* hv_out = ps_out + (size_t)NSn * 3;
  float* pv_out = hv_out + (size_t)NVn * H;
  hipLaunchKernelGGL(node_finalize, dim3(NSn), dim3(128), 0, stream,
                     h_s, p_s, aggS, cntS, dS, W(37), W(38), W(39), W(40), hs_out, ps_out);
  hipLaunchKernelGGL(node_finalize, dim3(NVn), dim3(128), 0, stream,
                     h_v, p_v, aggV, cntV, dV, W(41), W(42), W(43), W(44), hv_out, pv_out);
}
